// VectorQuantizer_90082644067050
// MI455X (gfx1250) — matrix operand order repaired and checked by reading, not compiled, not run on hardware
//
#include <hip/hip_runtime.h>
#include <hip/hip_bf16.h>
#include <math.h>

// Problem constants
#define Bd 16
#define Dd 256
#define Td 1024
#define Kd 8192
#define NROWS (Bd * Td)           // 16384
#define ZQ_ELEMS (Bd * Dd * Td)   // 4194304
#define LOSS_OFF ZQ_ELEMS
#define PERP_OFF (ZQ_ELEMS + 1)
#define IDX_OFF  (ZQ_ELEMS + 2)

typedef __attribute__((ext_vector_type(16))) __bf16    v16bf;
typedef __attribute__((ext_vector_type(8)))  float     v8f;
typedef __attribute__((ext_vector_type(4)))  float     vf4;
typedef __attribute__((ext_vector_type(4)))  unsigned  v4u;
typedef __attribute__((ext_vector_type(8)))  int       v8i;
typedef __attribute__((ext_vector_type(4)))  int       v4i;

union FragPack {
    struct { vf4 lo, hi; } f;
    v16bf v;
};

__device__ __forceinline__ unsigned short f2bf(float x) {
    unsigned u = __float_as_uint(x);
    unsigned r = u + 0x7fffu + ((u >> 16) & 1u);   // round-to-nearest-even
    return (unsigned short)(r >> 16);
}

// ---------------------------------------------------------------- init
__global__ void vq_init_kernel(float* acc) {
    if (threadIdx.x < 2) acc[threadIdx.x] = 0.0f;
}

// -------------------------------------------------- z [B,D,T] -> zbf16 [(b,t),d]
__launch_bounds__(256)
__global__ void vq_zt_kernel(const float* __restrict__ z,
                             unsigned short* __restrict__ zbf) {
    __shared__ float tile[32][33];
    const int bi  = blockIdx.x;        // 4096 blocks
    const int b   = bi >> 8;
    const int rem = bi & 255;
    const int d0  = (rem >> 5) << 5;   // d-tile * 32
    const int t0  = (rem & 31) << 5;   // t-tile * 32
    const int i   = threadIdx.x >> 5;  // 0..7
    const int j   = threadIdx.x & 31;  // 0..31
    const float* src = z + (size_t)b * Dd * Td;
#pragma unroll
    for (int r = 0; r < 4; r++) {
        int dl = i + r * 8;
        tile[dl][j] = src[(size_t)(d0 + dl) * Td + t0 + j];
    }
    __syncthreads();
#pragma unroll
    for (int r = 0; r < 4; r++) {
        int tl = i + r * 8;
        zbf[(size_t)(b * Td + t0 + tl) * Dd + d0 + j] = f2bf(tile[j][tl]);
    }
}

// -------------------------------------------- embedding -> bf16 + row norms
__launch_bounds__(256)
__global__ void vq_pe_kernel(const float* __restrict__ emb,
                             unsigned short* __restrict__ ebf,
                             float* __restrict__ enorm2) {
    __shared__ float red[256];
    const int k = blockIdx.x;         // 8192 blocks
    const int d = threadIdx.x;
    float v = emb[(size_t)k * Dd + d];
    ebf[(size_t)k * Dd + d] = f2bf(v);
    red[d] = v * v;
    __syncthreads();
#pragma unroll
    for (int s = 128; s > 0; s >>= 1) {
        if (d < s) red[d] += red[d + s];
        __syncthreads();
    }
    if (d == 0) enorm2[k] = red[0];
}

// -------------------------------------------------------- TDM chunk loader
// Issues a 1-D Tensor-Data-Mover DMA: 4096 x 8B = 32 KB (64 codes x 256 bf16)
// from global `src` into LDS byte-offset `ldsOff`.
// D# layout per CDNA5 ISA ch.8: group0 = {count/flags, lds_addr,
// global_addr[31:0], global_addr[56:32] | type=2<<30}; group1 carries
// data_size=8B (bits 17:16), tensor_dim0=4096 (bits 79:48),
// tile_dim0=4096 (bits 127:112); groups 2/3 unused (<=2-D tensor).
__device__ __forceinline__ void tdm_load_chunk(const unsigned short* src,
                                               unsigned ldsOff) {
    unsigned long long ga = (unsigned long long)(uintptr_t)src;
    v4u g0;
    g0.x = 1u;                                            // count=1, user desc
    g0.y = ldsOff;                                        // lds_addr (bytes)
    g0.z = (unsigned)(ga & 0xffffffffu);                  // global_addr[31:0]
    g0.w = (unsigned)((ga >> 32) & 0x01ffffffu) | (2u << 30);  // [56:32], type=2
    v8i g1;
    g1[0] = (int)(3u << 16);          // workgroup_mask=0, data_size=3 (8 bytes)
    g1[1] = (int)(0x1000u << 16);     // tensor_dim0[15:0] = 4096  (bits 63:48)
    g1[2] = 0;                        // tensor_dim0[31:16]=0, tensor_dim1 lo = 0
    g1[3] = (int)(0x1000u << 16);     // tile_dim0 = 4096          (bits 127:112)
    g1[4] = 0; g1[5] = 0; g1[6] = 0; g1[7] = 0;
    v4i zz4 = {0, 0, 0, 0};
    v8i zz8 = {0, 0, 0, 0, 0, 0, 0, 0};
    __builtin_amdgcn_tensor_load_to_lds(g0, g1, zz4, zz4, zz8, 0);
}

// ----------------------------------------------- fused WMMA GEMM + argmin
// block = 256 threads = 8 waves; each wave owns 16 rows; block covers 128 rows.
// Codebook streamed via TDM into double-buffered LDS (2 x 32 KB), issued by
// wave 0 and overlapped with WMMA compute; one barrier per 64-code chunk.
__launch_bounds__(256)
__global__ void vq_argmin_kernel(const unsigned short* __restrict__ zbf,
                                 const unsigned short* __restrict__ ebf,
                                 const float* __restrict__ enorm2,
                                 int* __restrict__ idxOut,
                                 float* __restrict__ idxOutF) {
    __shared__ alignas(16) unsigned short ldsB[2][64 * 256];  // 2 x 32 KB

    const int tid  = threadIdx.x;
    const int wave = tid >> 5;
    const int lane = tid & 31;
    const int n    = lane & 15;       // column within 16-wide tile / row M for A
    const int half = lane >> 4;
    const int rBase = blockIdx.x * 128 + wave * 16;

    const unsigned ldsOff0 = (unsigned)(uintptr_t)(&ldsB[0][0]);
    const unsigned ldsOff1 = (unsigned)(uintptr_t)(&ldsB[1][0]);

    // A fragments: 16 rows x 256 d, resident in VGPRs for the whole K loop.
    // ISA A-layout (16-bit, 16x32): v0..3 -> 8 consecutive bf16 at d = 32c + 8*half,
    //                               v4..7 -> 8 consecutive bf16 at d = 32c + 16 + 8*half.
    const vf4* zrow = (const vf4*)(zbf + (size_t)(rBase + n) * Dd);
    FragPack afr[8];
#pragma unroll
    for (int c = 0; c < 8; c++) {
        afr[c].f.lo = zrow[4 * c + half];
        afr[c].f.hi = zrow[4 * c + 2 + half];
    }

    float best[8];
    int   bidx[8];
#pragma unroll
    for (int i = 0; i < 8; i++) { best[i] = 3.0e38f; bidx[i] = 0; }

    // Prologue: kick off DMA of chunk 0 into buffer 0 (TDM ignores EXEC;
    // only wave 0 issues so the transfer happens once per workgroup).
    if (wave == 0) tdm_load_chunk(ebf, ldsOff0);

    const int NCHUNK = Kd / 64;   // 128
    for (int i = 0; i < NCHUNK; i++) {
        const int kc = i * 64;
        if (wave == 0) __builtin_amdgcn_s_wait_tensorcnt((short)0);  // chunk i landed
        __syncthreads();          // all waves: chunk i visible, chunk i-1 consumed
        if (wave == 0 && (i + 1) < NCHUNK) {
            // Overlap: DMA chunk i+1 into the other buffer while computing.
            tdm_load_chunk(ebf + (size_t)(kc + 64) * Dd,
                           ((i + 1) & 1) ? ldsOff1 : ldsOff0);
        }

        // Hoisted: batch the 4 code-norm loads for this chunk into one clause.
        float en[4];
#pragma unroll
        for (int ct = 0; ct < 4; ct++) en[ct] = enorm2[kc + ct * 16 + n];

        const unsigned short* buf = ldsB[i & 1];
#pragma unroll
        for (int ct = 0; ct < 4; ct++) {
            const int  col  = kc + ct * 16 + n;
            const vf4* brow = (const vf4*)(buf + (ct * 16 + n) * Dd);

            // Software pipeline: issue all 16 ds_load_b128 for this tile first,
            // so the WMMA chain runs behind partial s_wait_dscnt counts instead
            // of a full-latency wait per k-chunk.
            FragPack bfr[8];
#pragma unroll
            for (int c = 0; c < 8; c++) {
                // B-layout (32x16 bf16): v0..3 -> 8 bf16 at K = 8*half, v4..7 -> K = 16 + 8*half
                // -> two b128 LDS loads per half, matching the A layout.
                bfr[c].f.lo = brow[4 * c + half];
                bfr[c].f.hi = brow[4 * c + 2 + half];
            }

            v8f acc = {0.f, 0.f, 0.f, 0.f, 0.f, 0.f, 0.f, 0.f};
#pragma unroll
            for (int c = 0; c < 8; c++) {
                acc = __builtin_amdgcn_wmma_f32_16x16x32_bf16(
                        false, afr[c].v, false, bfr[c].v, (short)0, acc, false, false);
            }
#pragma unroll
            for (int q = 0; q < 8; q++) {
                float d = en[ct] - 2.0f * acc[q];  // ||e||^2 - 2 z.e (||z||^2 const)
                if (d < best[q]) { best[q] = d; bidx[q] = col; }
            }
        }
    }

    // C-layout: vgpr i, lanes [half*16 .. half*16+15] hold row (i + 8*half),
    // columns n = 0..15. Reduce across those 16 lanes.
#pragma unroll
    for (int i = 0; i < 8; i++) {
#pragma unroll
        for (int off = 8; off >= 1; off >>= 1) {
            float ob = __shfl_xor(best[i], off, 32);
            int   oi = __shfl_xor(bidx[i], off, 32);
            if (ob < best[i] || (ob == best[i] && oi < bidx[i])) {
                best[i] = ob; bidx[i] = oi;
            }
        }
        if (n == 0) {
            int row = rBase + i + half * 8;
            idxOut[row]  = bidx[i];
            idxOutF[row] = (float)bidx[i];
        }
    }
}

// ------------------------------------------- gather z_q + fused MSE loss
__launch_bounds__(256)
__global__ void vq_gather_loss_kernel(const float* __restrict__ z,
                                      const float* __restrict__ emb,
                                      const int* __restrict__ idxw,
                                      float* __restrict__ zq,
                                      float* __restrict__ lossAcc) {
    __shared__ float red[256];
    const int e  = blockIdx.x * 256 + threadIdx.x;    // < 4194304
    const int b  = e / (Dd * Td);
    const int r  = e % (Dd * Td);
    const int dd = r / Td;
    const int t  = r % Td;
    const int k  = idxw[b * Td + t];
    float q = emb[(size_t)k * Dd + dd];
    zq[e] = q;
    float df = z[e] - q;
    red[threadIdx.x] = df * df;
    __syncthreads();
#pragma unroll
    for (int s = 128; s > 0; s >>= 1) {
        if (threadIdx.x < s) red[threadIdx.x] += red[threadIdx.x + s];
        __syncthreads();
    }
    if (threadIdx.x == 0) atomicAdd(lossAcc, red[0]);
}

// ------------------------------------------------------------ perplexity
__launch_bounds__(256)
__global__ void vq_perp_kernel(const int* __restrict__ idxw,
                               float* __restrict__ perpAcc) {
    const int t = blockIdx.x * 256 + threadIdx.x;
    if (t >= Td) return;
    int v[Bd];
#pragma unroll
    for (int b = 0; b < Bd; b++) v[b] = idxw[b * Td + t];
    float s = 0.0f;
#pragma unroll
    for (int b = 0; b < Bd; b++) {
        int c = 0, first = -1;
#pragma unroll
        for (int b2 = 0; b2 < Bd; b2++) {
            if (v[b2] == v[b]) { c++; if (first < 0) first = b2; }
        }
        if (first == b) {
            float p = (float)c * (1.0f / (float)Bd);
            s += -(p * logf(p + 1e-10f));
        }
    }
    atomicAdd(perpAcc, s);
}

// -------------------------------------------------------------- finalize
__global__ void vq_fin_kernel(const float* __restrict__ acc,
                              float* __restrict__ outLoss,
                              float* __restrict__ outPerp) {
    if (threadIdx.x == 0) {
        // codebook_loss + 0.25 * commitment_loss = 1.25 * mean((z - z_q)^2)
        outLoss[0] = acc[0] * 1.25f / (float)ZQ_ELEMS;
        outPerp[0] = expf(acc[1]);
    }
}

extern "C" void kernel_launch(void* const* d_in, const int* in_sizes, int n_in,
                              void* d_out, int out_size, void* d_ws, size_t ws_size,
                              hipStream_t stream) {
    const float* z   = (const float*)d_in[0];   // [B, D, T]
    const float* emb = (const float*)d_in[1];   // [K, D]
    float* out = (float*)d_out;

    char* ws = (char*)d_ws;
    size_t off = 0;
    unsigned short* zbf = (unsigned short*)(ws + off); off += (size_t)NROWS * Dd * 2;  // 8 MB
    unsigned short* ebf = (unsigned short*)(ws + off); off += (size_t)Kd * Dd * 2;     // 4 MB
    float* enorm2 = (float*)(ws + off); off += (size_t)Kd * 4;
    int*   idxw   = (int*)(ws + off);   off += (size_t)NROWS * 4;
    float* acc    = (float*)(ws + off); off += 256;   // acc[0]=loss sum, acc[1]=entropy sum

    float* zq    = out;
    float* lossp = out + LOSS_OFF;
    float* perpp = out + PERP_OFF;
    float* idxf  = out + IDX_OFF;

    vq_init_kernel<<<1, 32, 0, stream>>>(acc);
    vq_zt_kernel<<<Bd * (Dd / 32) * (Td / 32), 256, 0, stream>>>(z, zbf);
    vq_pe_kernel<<<Kd, 256, 0, stream>>>(emb, ebf, enorm2);
    vq_argmin_kernel<<<NROWS / 128, 256, 0, stream>>>(zbf, ebf, enorm2, idxw, idxf);
    vq_gather_loss_kernel<<<ZQ_ELEMS / 256, 256, 0, stream>>>(z, emb, idxw, zq, acc + 0);
    vq_perp_kernel<<<(Td + 255) / 256, 256, 0, stream>>>(idxw, acc + 1);
    vq_fin_kernel<<<1, 32, 0, stream>>>(acc, lossp, perpp);
}
